// RNNWithAttention_75350906241537
// MI455X (gfx1250) — matrix operand order repaired and checked by reading, not compiled, not run on hardware
//
#include <hip/hip_runtime.h>
#include <hip/hip_bf16.h>

// ---------------------------------------------------------------------------
// Problem constants (match reference)
// ---------------------------------------------------------------------------
constexpr int Vn = 8192;
constexpr int En = 64;
constexpr int Hn = 256;
constexpr int Bn = 4;
constexpr int Tn = 512;
constexpr int Mrows = Bn * Tn;           // 2048

typedef __bf16 v16bf __attribute__((ext_vector_type(16)));
typedef __bf16 v8bf  __attribute__((ext_vector_type(8)));
typedef __bf16 v4bf  __attribute__((ext_vector_type(4)));
typedef float  v8f   __attribute__((ext_vector_type(8)));

__device__ __forceinline__ v16bf cat8(v8bf lo, v8bf hi) {
    return __builtin_shufflevector(lo, hi,
             0,1,2,3,4,5,6,7,8,9,10,11,12,13,14,15);
}

__device__ __forceinline__ float fast_tanh(float x) {
    x = fminf(15.0f, fmaxf(-15.0f, x));
    float e = __expf(2.0f * x);
    return (e - 1.0f) / (e + 1.0f);
}

// ---------------------------------------------------------------------------
// f32 -> bf16 bulk conversion (n % 4 == 0)
// ---------------------------------------------------------------------------
__global__ void cvt_f32_bf16_kernel(const float* __restrict__ in,
                                    __bf16* __restrict__ out, int n) {
    int i = (blockIdx.x * blockDim.x + threadIdx.x) * 4;
    if (i < n) {
        float4 v = *(const float4*)(in + i);
        v4bf o;
        o[0] = (__bf16)v.x; o[1] = (__bf16)v.y;
        o[2] = (__bf16)v.z; o[3] = (__bf16)v.w;
        *(v4bf*)(out + i) = o;
    }
}

// ---------------------------------------------------------------------------
// Embedding gather: e[b,t,:] = emb[x[b,t],:]
// ---------------------------------------------------------------------------
__global__ void embed_kernel(const int* __restrict__ x,
                             const float* __restrict__ emb,
                             float* __restrict__ e) {
    int idx = blockIdx.x * blockDim.x + threadIdx.x;    // over B*T*E
    if (idx < Mrows * En) {
        int bt = idx >> 6;          // /64
        int j  = idx & (En - 1);
        e[idx] = emb[(size_t)x[bt] * En + j];
    }
}

// ---------------------------------------------------------------------------
// GEMM: C[m,n] = sum_k A[m,k] * W[n,k] + bias0[n] + bias1[n]
// bf16 WMMA 16x16x32, f32 accumulate. One wave computes a 32(M) x 64(N)
// strip: 2 M-tiles x 4 N-tiles; each B fragment feeds two WMMAs.
// M%32==0, N%64==0, K%32==0.
// ---------------------------------------------------------------------------
__global__ void gemm_bf16_wmma(const __bf16* __restrict__ A,
                               const __bf16* __restrict__ W,
                               const float* __restrict__ bias0,
                               const float* __restrict__ bias1,
                               float* __restrict__ C,
                               int M, int N, int K) {
    const int wavesPerBlock = blockDim.x >> 5;
    const int waveId = blockIdx.x * wavesPerBlock + (threadIdx.x >> 5);
    const int lane = threadIdx.x & 31;
    const int nGroups = N >> 6;                 // groups of 4 n-tiles
    const int totalWaves = (M >> 5) * nGroups;
    if (waveId >= totalWaves) return;           // wave-uniform: EXEC stays all-1
    const int mPair  = waveId / nGroups;
    const int nGroup = waveId % nGroups;

    const int row  = lane & 15;                 // M index for A, N index for B/D
    const int half = lane >> 4;

    v8f acc0[4] = {};
    v8f acc1[4] = {};
    const __bf16* arow0 = A + (size_t)(mPair * 32 + row) * K;
    const __bf16* arow1 = arow0 + (size_t)16 * K;
    const __bf16* brow0 = W + (size_t)((nGroup << 6) + row) * K;

    for (int kb = 0; kb < K; kb += 32) {
        if (kb + 32 < K) {
            __builtin_prefetch(arow0 + kb + 32, 0, 2);
            __builtin_prefetch(arow1 + kb + 32, 0, 2);
        }
        // A fragments: lane = M row; halves carry K 0-15 vs 16-31
        v16bf af0 = cat8(*(const v8bf*)(arow0 + kb + 16 * half),
                         *(const v8bf*)(arow0 + kb + 8 + 16 * half));
        v16bf af1 = cat8(*(const v8bf*)(arow1 + kb + 16 * half),
                         *(const v8bf*)(arow1 + kb + 8 + 16 * half));
        // B fragments: lane = N column; halves carry K 0-15 vs 16-31
        #pragma unroll
        for (int t = 0; t < 4; ++t) {
            const __bf16* bp = brow0 + (size_t)t * 16 * K + kb + 16 * half;
            v16bf bfr = cat8(*(const v8bf*)bp, *(const v8bf*)(bp + 8));
            acc0[t] = __builtin_amdgcn_wmma_f32_16x16x32_bf16(
                false, af0, false, bfr, (short)0, acc0[t], false, false);
            acc1[t] = __builtin_amdgcn_wmma_f32_16x16x32_bf16(
                false, af1, false, bfr, (short)0, acc1[t], false, false);
        }
    }

    // store: D layout lane = N, VGPR r -> M = r + 8*half
    #pragma unroll
    for (int t = 0; t < 4; ++t) {
        int n = (nGroup << 6) + (t << 4) + (lane & 15);
        float bs = (bias0 ? bias0[n] : 0.0f) + (bias1 ? bias1[n] : 0.0f);
        #pragma unroll
        for (int r = 0; r < 8; ++r) {
            int m0 = mPair * 32 + r + 8 * half;
            C[(size_t)m0 * N + n]        = acc0[t][r] + bs;
            C[(size_t)(m0 + 16) * N + n] = acc1[t][r] + bs;
        }
    }
}

// ---------------------------------------------------------------------------
// Async copy of one 16-timestep xp chunk (4 batches x 16 t x 256 h, f32)
// into LDS via CDNA5 global_load_async_to_lds_b128. 8 issues per lane.
// ---------------------------------------------------------------------------
__device__ __forceinline__ void issue_xp_chunk(const float* __restrict__ xp,
                                               float* xpsDst, int c, int tid) {
    if (c * 16 >= Tn) return;
    for (int p = tid; p < 4096; p += 512) {     // 4096 float4 packets
        int b   = p >> 10;
        int rem = p & 1023;
        const float* g = xp + ((size_t)b * Tn + (size_t)c * 16) * Hn + (size_t)rem * 4;
        auto lp = (__attribute__((address_space(3))) float*)(xpsDst + b * 4096 + rem * 4);
        asm volatile("global_load_async_to_lds_b128 %0, %1, off"
                     :: "v"(lp), "v"(g) : "memory");
    }
}

// ---------------------------------------------------------------------------
// Sequential RNN layer on the matrix pipe. xp[b,t,h] holds x_t@W_ih^T + both
// biases. Recurrence h_new = tanh(xp_t + h @ W_hh^T) runs as 16 N-tiles of
// v_wmma_f32_16x16x32_bf16 (one wave each; M=16 with rows 4..15 as don't-care
// padding). Each wave keeps its whole W_hh tile resident in 64 VGPRs of B
// fragments for all 512 steps. h lives in LDS as bf16 in A-fragment-friendly
// row-major layout. xp is double-buffered into LDS with async loads.
// ---------------------------------------------------------------------------
__global__ void rnn_layer_wmma_kernel(const float* __restrict__ xp,
                                      const __bf16* __restrict__ Whh_bf,
                                      float* __restrict__ y,
                                      float* __restrict__ hT) {
    extern __shared__ char smemraw[];
    float*  xps = (float*)smemraw;                          // 2 x 16384 f32
    __bf16* hbf = (__bf16*)(smemraw + 2 * 16384 * sizeof(float)); // [16][256]

    const int tid  = threadIdx.x;       // blockDim = 512 (16 waves)
    const int w    = tid >> 5;          // wave id == N-tile id (0..15)
    const int lane = tid & 31;
    const int row  = lane & 15;
    const int half = lane >> 4;
    const int n    = (w << 4) + row;    // output unit handled by this lane col

    // ---- resident B fragments: lane = N col; halves carry K 0-15 / 16-31
    v16bf barr[8];
    #pragma unroll
    for (int kb8 = 0; kb8 < 8; ++kb8) {
        const __bf16* bp = Whh_bf + (size_t)n * Hn + kb8 * 32 + 16 * half;
        barr[kb8] = cat8(*(const v8bf*)bp, *(const v8bf*)(bp + 8));
    }

    // ---- init h (incl. pad rows) to zero
    for (int i = tid; i < 16 * Hn; i += 512) hbf[i] = (__bf16)0.0f;

    issue_xp_chunk(xp, xps,         0, tid);
    issue_xp_chunk(xp, xps + 16384, 1, tid);
    __syncthreads();

    for (int t = 0; t < Tn; ++t) {
        const int tt  = t & 15;
        const int c   = t >> 4;
        const int buf = c & 1;
        if (tt == 0) {
            if (c + 1 < Tn / 16) {
                asm volatile("s_wait_asynccnt 0x8" ::: "memory");  // chunk c done
            } else {
                asm volatile("s_wait_asynccnt 0x0" ::: "memory");  // last chunk
            }
            __syncthreads();
        }

        // ---- h @ W_hh^T for this N-tile: 8 chained WMMAs, A from LDS h
        v8f acc = {};
        #pragma unroll
        for (int kb8 = 0; kb8 < 8; ++kb8) {
            const __bf16* ap = hbf + row * Hn + kb8 * 32 + 16 * half;
            v16bf af = cat8(*(const v8bf*)ap, *(const v8bf*)(ap + 8));
            acc = __builtin_amdgcn_wmma_f32_16x16x32_bf16(
                false, af, false, barr[kb8], (short)0, acc, false, false);
        }
        __syncthreads();                       // all waves done reading hbf

        // ---- D layout: lane = N, VGPR r -> M = r + 8*half; rows 0..3 valid
        if (half == 0) {
            const float* xpt = xps + buf * 16384 + tt * Hn;
            #pragma unroll
            for (int r = 0; r < Bn; ++r) {
                float hn = fast_tanh(acc[r] + xpt[r * 4096 + n]);
                y[((size_t)r * Tn + t) * Hn + n] = hn;
                hbf[r * Hn + n] = (__bf16)hn;
                if (t == Tn - 1) hT[r * Hn + n] = hn;
            }
        }
        __syncthreads();                       // h updated, xps chunk consumed
        if (tt == 15) issue_xp_chunk(xp, xps + buf * 16384, c + 2, tid);
    }
}

// ---------------------------------------------------------------------------
// Bahdanau attention + causal softmax + context. One block per (b, i).
// ---------------------------------------------------------------------------
__global__ void attention_kernel(const float* __restrict__ q,
                                 const float* __restrict__ k,
                                 const float* __restrict__ y1,
                                 const float* __restrict__ vvec,
                                 float* __restrict__ ctx) {
    __shared__ float qs[Hn];
    __shared__ float vs[Hn];
    __shared__ float sc[Tn];
    __shared__ float red[512];

    const int b = blockIdx.x / Tn;
    const int i = blockIdx.x % Tn;
    const int tid = threadIdx.x;          // blockDim = 512

    for (int h = tid; h < Hn; h += blockDim.x) {
        qs[h] = q[((size_t)b * Tn + i) * Hn + h];
        vs[h] = vvec[h];
    }
    __syncthreads();

    float s = -1e30f;
    if (tid <= i) {
        const float* kr = k + ((size_t)b * Tn + tid) * Hn;
        float a = 0.0f;
        #pragma unroll 4
        for (int h = 0; h < Hn; ++h) a += vs[h] * fast_tanh(qs[h] + kr[h]);
        s = a;
    }
    red[tid] = s;
    __syncthreads();
    for (int off = 256; off > 0; off >>= 1) {
        if (tid < off) red[tid] = fmaxf(red[tid], red[tid + off]);
        __syncthreads();
    }
    float mx = red[0];
    __syncthreads();

    float ev = (tid <= i) ? __expf(s - mx) : 0.0f;
    sc[tid]  = ev;
    red[tid] = ev;
    __syncthreads();
    for (int off = 256; off > 0; off >>= 1) {
        if (tid < off) red[tid] += red[tid + off];
        __syncthreads();
    }
    float inv = 1.0f / red[0];
    __syncthreads();

    for (int h = tid; h < Hn; h += blockDim.x) {
        float c = 0.0f;
        for (int j = 0; j <= i; ++j)
            c += sc[j] * y1[((size_t)b * Tn + j) * Hn + h];
        ctx[((size_t)b * Tn + i) * Hn + h] = c * inv;
    }
}

// ---------------------------------------------------------------------------
// LayerNorm over cat([y1, ctx], -1) (2H = 512). One block (256 thr) per row.
// ---------------------------------------------------------------------------
__global__ void layernorm_cat_kernel(const float* __restrict__ y1,
                                     const float* __restrict__ ctx,
                                     const float* __restrict__ gamma,
                                     const float* __restrict__ beta,
                                     float* __restrict__ out) {
    __shared__ float red[256];
    const int row = blockIdx.x;
    const int tid = threadIdx.x;

    float a = y1[(size_t)row * Hn + tid];
    float c = ctx[(size_t)row * Hn + tid];

    red[tid] = a + c;
    __syncthreads();
    for (int off = 128; off > 0; off >>= 1) {
        if (tid < off) red[tid] += red[tid + off];
        __syncthreads();
    }
    float mu = red[0] * (1.0f / (2 * Hn));
    __syncthreads();

    float da = a - mu, dc = c - mu;
    red[tid] = da * da + dc * dc;
    __syncthreads();
    for (int off = 128; off > 0; off >>= 1) {
        if (tid < off) red[tid] += red[tid + off];
        __syncthreads();
    }
    float rstd = rsqrtf(red[0] * (1.0f / (2 * Hn)) + 1e-5f);

    out[(size_t)row * (2 * Hn) + tid]      = da * rstd * gamma[tid] + beta[tid];
    out[(size_t)row * (2 * Hn) + Hn + tid] = dc * rstd * gamma[Hn + tid] + beta[Hn + tid];
}

// ---------------------------------------------------------------------------
// Host-side launch
// ---------------------------------------------------------------------------
extern "C" void kernel_launch(void* const* d_in, const int* in_sizes, int n_in,
                              void* d_out, int out_size, void* d_ws, size_t ws_size,
                              hipStream_t stream) {
    (void)in_sizes; (void)n_in; (void)out_size; (void)ws_size;

    const int*   x     = (const int*)  d_in[0];
    const float* emb   = (const float*)d_in[1];
    const float* W_ih0 = (const float*)d_in[2];
    const float* W_hh0 = (const float*)d_in[3];
    const float* b_ih0 = (const float*)d_in[4];
    const float* b_hh0 = (const float*)d_in[5];
    const float* W_ih1 = (const float*)d_in[6];
    const float* W_hh1 = (const float*)d_in[7];
    const float* b_ih1 = (const float*)d_in[8];
    const float* b_hh1 = (const float*)d_in[9];
    const float* Wq    = (const float*)d_in[10];
    const float* Wk    = (const float*)d_in[11];
    const float* vvec  = (const float*)d_in[12];
    const float* gamma = (const float*)d_in[13];
    const float* beta  = (const float*)d_in[14];
    const float* Wfc   = (const float*)d_in[15];
    const float* bfc   = (const float*)d_in[16];

    float* logits = (float*)d_out;                       // (B,T,V)
    float* hout   = logits + (size_t)Mrows * Vn;         // (L,B,H)

    // ---- f32 workspace ----
    float* ws  = (float*)d_ws;
    float* e   = ws;                  // 2048*64
    float* xp  = e   + Mrows * En;    // 2048*256 (reused for both layers)
    float* y0  = xp  + Mrows * Hn;
    float* y1  = y0  + Mrows * Hn;
    float* q   = y1  + Mrows * Hn;
    float* kk  = q   + Mrows * Hn;
    float* ctx = kk  + Mrows * Hn;
    float* nrm = ctx + Mrows * Hn;    // 2048*512
    float* fend = nrm + Mrows * 2 * Hn;

    // ---- bf16 workspace ----
    __bf16* bws     = (__bf16*)fend;
    __bf16* e_bf    = bws;                       // 2048*64
    __bf16* a_bf    = e_bf  + Mrows * En;        // 2048*512 (y0/y1/nrm reuse)
    __bf16* Wih0_bf = a_bf  + Mrows * 2 * Hn;    // 256*64
    __bf16* Wih1_bf = Wih0_bf + Hn * En;         // 256*256
    __bf16* Wq_bf   = Wih1_bf + Hn * Hn;
    __bf16* Wk_bf   = Wq_bf   + Hn * Hn;
    __bf16* Whh0_bf = Wk_bf   + Hn * Hn;
    __bf16* Whh1_bf = Whh0_bf + Hn * Hn;
    __bf16* Wfc_bf  = Whh1_bf + Hn * Hn;         // 8192*512

    auto cvt = [&](const float* in, __bf16* out, int n) {
        cvt_f32_bf16_kernel<<<(n / 4 + 255) / 256, 256, 0, stream>>>(in, out, n);
    };
    auto gemm = [&](const __bf16* A, const __bf16* W, const float* bb0,
                    const float* bb1, float* C, int M, int N, int K) {
        int totalWaves = (M / 32) * (N / 64);
        int blocks = (totalWaves + 7) / 8;
        gemm_bf16_wmma<<<blocks, 256, 0, stream>>>(A, W, bb0, bb1, C, M, N, K);
    };

    size_t rnnShmem = 2 * 16384 * sizeof(float) + 16 * Hn * sizeof(__bf16); // 136KB
    hipFuncSetAttribute((const void*)rnn_layer_wmma_kernel,
                        hipFuncAttributeMaxDynamicSharedMemorySize,
                        (int)rnnShmem);

    // 0) weight conversions (independent of everything else)
    cvt(W_ih0, Wih0_bf, Hn * En);
    cvt(W_ih1, Wih1_bf, Hn * Hn);
    cvt(Wq,    Wq_bf,   Hn * Hn);
    cvt(Wk,    Wk_bf,   Hn * Hn);
    cvt(W_hh0, Whh0_bf, Hn * Hn);
    cvt(W_hh1, Whh1_bf, Hn * Hn);
    cvt(Wfc,   Wfc_bf,  Vn * 2 * Hn);

    // 1) embedding
    embed_kernel<<<(Mrows * En + 255) / 256, 256, 0, stream>>>(x, emb, e);
    cvt(e, e_bf, Mrows * En);

    // 2) layer 0: input projection (parallel), then sequential recurrence
    gemm(e_bf, Wih0_bf, b_ih0, b_hh0, xp, Mrows, Hn, En);
    rnn_layer_wmma_kernel<<<1, 512, rnnShmem, stream>>>(xp, Whh0_bf, y0, hout);

    // 3) layer 1
    cvt(y0, a_bf, Mrows * Hn);
    gemm(a_bf, Wih1_bf, b_ih1, b_hh1, xp, Mrows, Hn, Hn);
    rnn_layer_wmma_kernel<<<1, 512, rnnShmem, stream>>>(xp, Whh1_bf, y1, hout + Bn * Hn);

    // 4) attention projections (share one bf16 copy of y1)
    cvt(y1, a_bf, Mrows * Hn);
    gemm(a_bf, Wq_bf, nullptr, nullptr, q,  Mrows, Hn, Hn);
    gemm(a_bf, Wk_bf, nullptr, nullptr, kk, Mrows, Hn, Hn);

    // 5) additive attention + causal softmax + context
    attention_kernel<<<Mrows, 512, 0, stream>>>(q, kk, y1, vvec, ctx);

    // 6) layernorm over concat
    layernorm_cat_kernel<<<Mrows, 256, 0, stream>>>(y1, ctx, gamma, beta, nrm);

    // 7) final FC -> logits (dominant GEMM: 2048 x 8192 x 512)
    cvt(nrm, a_bf, Mrows * 2 * Hn);
    gemm(a_bf, Wfc_bf, bfc, nullptr, logits, Mrows, Vn, 2 * Hn);
}
